// AlloLayer_60035052863916
// MI455X (gfx1250) — matrix operand order repaired and checked by reading, not compiled, not run on hardware
//
#include <hip/hip_runtime.h>
#include <math.h>

// Problem constants (from the reference): B*T=16384 rows, C=512 phones,
// A=4096 arcs, P=256 phonemes.
#define CDIM 512
#define PDIM 256
#define NARCS 4096
#define ROWS_PER_WG 128   // 8 waves * 16 rows
#define NWAVES 8

typedef _Float16 v16h  __attribute__((ext_vector_type(16)));
typedef _Float16 half8 __attribute__((ext_vector_type(8)));
typedef float    v8f   __attribute__((ext_vector_type(8)));

// ---------------------------------------------------------------------------
// Prep kernels: build Mt[P][C] (transposed weight matrix) = scatter of
// exp(alloW[a]) into (phoneme(a), phone(a)); f32 stage -> f16 for WMMA B.
// ---------------------------------------------------------------------------
__global__ void zero_ws(float* __restrict__ p, int n) {
    int i = blockIdx.x * blockDim.x + threadIdx.x;
    if (i < n) p[i] = 0.0f;
}

__global__ void build_Mt(const float* __restrict__ alloW,
                         const int* __restrict__ phone,
                         const int* __restrict__ phoneme,
                         float* __restrict__ MtF32) {
    int a = blockIdx.x * blockDim.x + threadIdx.x;
    if (a < NARCS) {
        atomicAdd(&MtF32[phoneme[a] * CDIM + phone[a]], __expf(alloW[a]));
    }
}

__global__ void cvt_Mt(const float* __restrict__ src, _Float16* __restrict__ dst, int n) {
    int i = blockIdx.x * blockDim.x + threadIdx.x;
    if (i < n) dst[i] = (_Float16)src[i];
}

// ---------------------------------------------------------------------------
// Fused softmax + GEMM + redistribute/log kernel.
// Each workgroup: 8 waves, each wave owns a 16-row stripe.
// Phase 1: softmax(hs row) -> f16 probs in LDS (128 rows x 512 halves = 128KB).
// Phase 2: per wave, D[16x256] = A[16x512] x Mt^T via v_wmma_f32_16x16x32_f16,
//          K tiled x16, N tiled x16, accumulators held in VGPRs.
// Phase 3: row sums from accumulators, out = log(acc - (rowsum-1)/P).
// ---------------------------------------------------------------------------
__launch_bounds__(NWAVES * 32)
__global__ void allo_fused(const float* __restrict__ hs,
                           const _Float16* __restrict__ Mt,
                           float* __restrict__ out) {
    extern __shared__ _Float16 ldsA[];        // [ROWS_PER_WG][CDIM]
    const int lane    = threadIdx.x & 31;
    const int wave    = threadIdx.x >> 5;
    const int rowBase = blockIdx.x * ROWS_PER_WG;
    const int waveRow = wave * 16;

    // ---- Phase 1: softmax, one row at a time per wave (32 lanes x 16 elems)
    for (int i = 0; i < 16; ++i) {
        const int lrow = waveRow + i;
        const float4* rp = (const float4*)(hs + (size_t)(rowBase + lrow) * CDIM);
        float4 x[4];
        #pragma unroll
        for (int j = 0; j < 4; ++j) x[j] = rp[lane + j * 32];

        float m = x[0].x;
        #pragma unroll
        for (int j = 0; j < 4; ++j) {
            m = fmaxf(m, x[j].x); m = fmaxf(m, x[j].y);
            m = fmaxf(m, x[j].z); m = fmaxf(m, x[j].w);
        }
        #pragma unroll
        for (int mask = 16; mask >= 1; mask >>= 1)
            m = fmaxf(m, __shfl_xor(m, mask, 32));

        float s = 0.0f;
        #pragma unroll
        for (int j = 0; j < 4; ++j) {
            x[j].x = __expf(x[j].x - m); s += x[j].x;
            x[j].y = __expf(x[j].y - m); s += x[j].y;
            x[j].z = __expf(x[j].z - m); s += x[j].z;
            x[j].w = __expf(x[j].w - m); s += x[j].w;
        }
        #pragma unroll
        for (int mask = 16; mask >= 1; mask >>= 1)
            s += __shfl_xor(s, mask, 32);
        const float inv = __frcp_rn(s);

        _Float16* dp = ldsA + (size_t)lrow * CDIM;
        #pragma unroll
        for (int j = 0; j < 4; ++j) {
            const int col = (lane + j * 32) * 4;
            dp[col + 0] = (_Float16)(x[j].x * inv);
            dp[col + 1] = (_Float16)(x[j].y * inv);
            dp[col + 2] = (_Float16)(x[j].z * inv);
            dp[col + 3] = (_Float16)(x[j].w * inv);
        }
    }
    __syncthreads();

    // ---- Phase 2: WMMA GEMM. acc[nt] covers N = nt*16 .. nt*16+15.
    const v8f vzero = {0.f, 0.f, 0.f, 0.f, 0.f, 0.f, 0.f, 0.f};
    v8f acc[16];
    #pragma unroll
    for (int nt = 0; nt < 16; ++nt) acc[nt] = vzero;

    const int nLane  = lane & 15;   // A: row m = lane&15;  B/D: column n = lane&15
    const int hiHalf = lane >> 4;   // upper half-wave selects the second K run
    const _Float16* aRow = ldsA + (size_t)(waveRow + nLane) * CDIM;
    const _Float16* bCol = Mt + (size_t)nLane * CDIM;

    #pragma unroll
    for (int kt = 0; kt < 16; ++kt) {
        const int k0 = kt * 32;
        // A fragment (16x32 f16): lane<16 holds K {k0..k0+7, k0+16..k0+23},
        // lane>=16 holds K {k0+8..k0+15, k0+24..k0+31}. Two 16B LDS loads.
        const _Float16* ap = aRow + k0 + hiHalf * 8;
        half8 alo = *(const half8*)(ap);
        half8 ahi = *(const half8*)(ap + 16);
        v16h afrag = __builtin_shufflevector(alo, ahi,
            0, 1, 2, 3, 4, 5, 6, 7, 8, 9, 10, 11, 12, 13, 14, 15);
        #pragma unroll
        for (int nt = 0; nt < 16; ++nt) {
            // B fragment (32x16 f16): lane = column n; lanes 0-15 hold K {k0..k0+7, k0+16..k0+23},
            // lanes 16-31 hold K {k0+8..k0+15, k0+24..k0+31} in Mt[P][C]. Two 16B loads.
            const _Float16* bp = bCol + (size_t)(nt * 16) * CDIM + k0 + hiHalf * 8;
            half8 blo = *(const half8*)(bp);
            half8 bhi = *(const half8*)(bp + 16);
            v16h bfrag = __builtin_shufflevector(blo, bhi,
                0, 1, 2, 3, 4, 5, 6, 7, 8, 9, 10, 11, 12, 13, 14, 15);
            acc[nt] = __builtin_amdgcn_wmma_f32_16x16x32_f16(
                /*neg_a=*/false, afrag, /*neg_b=*/false, bfrag,
                /*c_mod=*/(short)0, acc[nt], /*reuse_a=*/false, /*reuse_b=*/false);
        }
    }

    // ---- Phase 3: row sums + redistribute + log.
    // C/D layout: lane L<16 -> n=L, m=vgpr; L>=16 -> n=L-16, m=vgpr+8.
    float ps[8];
    #pragma unroll
    for (int v = 0; v < 8; ++v) ps[v] = 0.0f;
    #pragma unroll
    for (int nt = 0; nt < 16; ++nt) {
        #pragma unroll
        for (int v = 0; v < 8; ++v) ps[v] += acc[nt][v];
    }
    // reduce over the 16 lanes that share each row-group
    #pragma unroll
    for (int mask = 1; mask < 16; mask <<= 1) {
        #pragma unroll
        for (int v = 0; v < 8; ++v) ps[v] += __shfl_xor(ps[v], mask, 16);
    }
    float corr[8];
    #pragma unroll
    for (int v = 0; v < 8; ++v) corr[v] = (ps[v] - 1.0f) * (1.0f / (float)PDIM);

    const int mOff = hiHalf * 8;
    float* outBase = out + (size_t)(rowBase + waveRow) * PDIM;
    #pragma unroll
    for (int nt = 0; nt < 16; ++nt) {
        const int n = nt * 16 + nLane;
        #pragma unroll
        for (int v = 0; v < 8; ++v) {
            outBase[(size_t)(mOff + v) * PDIM + n] = __logf(acc[nt][v] - corr[v]);
        }
    }
}

// ---------------------------------------------------------------------------
extern "C" void kernel_launch(void* const* d_in, const int* in_sizes, int n_in,
                              void* d_out, int out_size, void* d_ws, size_t ws_size,
                              hipStream_t stream) {
    const float* hs      = (const float*)d_in[0];
    const float* alloW   = (const float*)d_in[1];
    const int*   phone   = (const int*)d_in[2];
    const int*   phoneme = (const int*)d_in[3];
    float* outp = (float*)d_out;

    float*    MtF32 = (float*)d_ws;
    _Float16* MtH   = (_Float16*)((char*)d_ws + (size_t)CDIM * PDIM * sizeof(float));

    const int mElems = CDIM * PDIM;
    zero_ws<<<(mElems + 255) / 256, 256, 0, stream>>>(MtF32, mElems);
    build_Mt<<<(NARCS + 255) / 256, 256, 0, stream>>>(alloW, phone, phoneme, MtF32);
    cvt_Mt<<<(mElems + 255) / 256, 256, 0, stream>>>(MtF32, MtH, mElems);

    const int rows = in_sizes[0] / CDIM;          // B*T = 16384
    const int grid = rows / ROWS_PER_WG;          // 128 workgroups
    const size_t ldsBytes = (size_t)ROWS_PER_WG * CDIM * sizeof(_Float16); // 128 KB
    allo_fused<<<grid, NWAVES * 32, ldsBytes, stream>>>(hs, MtH, outp);
}
